// DI_MVS_28698971472076
// MI455X (gfx1250) — matrix operand order repaired and checked by reading, not compiled, not run on hardware
//
#include <hip/hip_runtime.h>
#include <hip/hip_bf16.h>

// Problem constants (from reference)
#define BB 2
#define VV 5
#define CC 32
#define HH 128
#define WW 160
#define DD 48
#define HW (HH * WW)
#define NSRC (VV - 1)

#define TILE 16          // ref pixels per tile (WMMA M dim)
#define QW_MAX 48        // max epipolar bbox width  (parallax ~11px + tile 16 + pad)
#define QH_MAX 20        // max epipolar bbox height (parallax ~11px + pad)
#define NQ_MAX (QW_MAX * QH_MAX)   // 960 usable corr columns
#define NQ_ROWS 976      // >= max padded qcol (959) + 1; row = one q, 16 floats (64B)

#define WS_BF16_OFFSET 512         // byte offset of bf16 channel-last features in d_ws

typedef __attribute__((ext_vector_type(16))) __bf16 v16bf;
typedef __attribute__((ext_vector_type(8)))  __bf16 v8bf;
typedef __attribute__((ext_vector_type(8)))  float  v8f;
typedef __attribute__((ext_vector_type(4)))  float  v4f;

static __device__ __forceinline__ int imin(int a, int b) { return a < b ? a : b; }
static __device__ __forceinline__ int imax(int a, int b) { return a > b ? a : b; }

static __device__ __forceinline__ __bf16 f2bf(float f) {
  union { float f; unsigned u; } a; a.f = f;
  unsigned u = a.u;
  u += 0x7fffu + ((u >> 16) & 1u);            // round-to-nearest-even
  union { unsigned short s; __bf16 b; } r; r.s = (unsigned short)(u >> 16);
  return r.b;
}

// ---------------- pass 0a: fold projections, ref inverse, per-view rot/trans --

static __device__ void mat3_from_fold(const float* pm, float* A, float* t) {
  // pm -> (2,4,4): E (extrinsic) then K (intrinsic). folded = [[K3*E3x3, K3*tE],[0,1]]
  const float* E = pm;
  const float* K = pm + 16;
  for (int i = 0; i < 3; ++i) {
    for (int j = 0; j < 3; ++j) {
      float s = 0.f;
      for (int k = 0; k < 3; ++k) s += K[i * 4 + k] * E[k * 4 + j];
      A[i * 3 + j] = s;
    }
    float s = 0.f;
    for (int k = 0; k < 3; ++k) s += K[i * 4 + k] * E[k * 4 + 3];
    t[i] = s;
  }
}

static __device__ void inv3(const float* A, float* Ai) {
  float a = A[0], b = A[1], c = A[2], d = A[3], e = A[4], f = A[5], g = A[6], h = A[7], i = A[8];
  float c0 = e * i - f * h, c1 = c * h - b * i, c2 = b * f - c * e;
  float c3 = f * g - d * i, c4 = a * i - c * g, c5 = c * d - a * f;
  float c6 = d * h - e * g, c7 = b * g - a * h, c8 = a * e - b * d;
  float rd = 1.0f / (a * c0 + b * c3 + c * c6);
  Ai[0] = c0 * rd; Ai[1] = c1 * rd; Ai[2] = c2 * rd;
  Ai[3] = c3 * rd; Ai[4] = c4 * rd; Ai[5] = c5 * rd;
  Ai[6] = c6 * rd; Ai[7] = c7 * rd; Ai[8] = c8 * rd;
}

__global__ void mvs_setup_kernel(const float* __restrict__ pm, float* __restrict__ ws) {
  if (threadIdx.x != 0 || blockIdx.x != 0) return;
  for (int b = 0; b < BB; ++b) {
    float Aref[9], tref[3], Ai[9];
    mat3_from_fold(pm + (b * VV + 0) * 32, Aref, tref);
    inv3(Aref, Ai);
    for (int v = 1; v < VV; ++v) {
      float Av[9], tv[3], R[9], T[3];
      mat3_from_fold(pm + (b * VV + v) * 32, Av, tv);
      for (int i = 0; i < 3; ++i)
        for (int j = 0; j < 3; ++j) {
          float s = 0.f;
          for (int k = 0; k < 3; ++k) s += Av[i * 3 + k] * Ai[k * 3 + j];
          R[i * 3 + j] = s;
        }
      for (int i = 0; i < 3; ++i) {
        float s = 0.f;
        for (int k = 0; k < 3; ++k) s += R[i * 3 + k] * tref[k];
        T[i] = tv[i] - s;      // folded_src @ ref_inv : translation column
      }
      float* o = ws + ((b * NSRC) + (v - 1)) * 12;
      for (int i = 0; i < 9; ++i) o[i] = R[i];
      for (int i = 0; i < 3; ++i) o[9 + i] = T[i];
    }
  }
}

// ---------------- pass 0b: f32 (B,V,C,H,W) -> bf16 channel-last (B,V,H,W,C) ---

__global__ __launch_bounds__(256) void mvs_convert_kernel(
    const float* __restrict__ feat, __bf16* __restrict__ featbf) {
  size_t pix = (size_t)blockIdx.x * 256 + threadIdx.x;   // over B*V*H*W
  if (pix >= (size_t)BB * VV * HW) return;
  size_t bv = pix / HW;
  size_t p  = pix - bv * HW;
  const float* src = feat + bv * (size_t)CC * HW + p;    // stride HW between channels
  __bf16* dst = featbf + pix * CC;                       // 64B contiguous per pixel
#pragma unroll
  for (int k = 0; k < 4; ++k) {
    v8bf w;
#pragma unroll
    for (int e = 0; e < 8; ++e) w[e] = f2bf(src[(size_t)(k * 8 + e) * HW]);
    *(v8bf*)(dst + k * 8) = w;
  }
}

// ---------------- main: corr GEMM (WMMA bf16) + bilinear blend -----------------

__global__ __launch_bounds__(256) void mvs_main_kernel(
    const __bf16* __restrict__ featbf,  // (B,V,H,W,C) bf16
    const float* __restrict__ dv,       // (B,D)
    const float* __restrict__ vw,       // (B,V-1,H,W)
    const float* __restrict__ rt,       // ws: (B,4,12) rot+trans
    float* __restrict__ out) {          // (B,1,D,H,W)
  __shared__ float corr[NQ_ROWS][TILE]; // 62.5 KB; row = one q column (64B, 16 ref px)
  __shared__ float sx[96], sy[96];
  __shared__ int sbox[6];               // qx0,qy0,qw,qh,nq,nchunk

  const int b    = blockIdx.z;
  const int y    = blockIdx.y;
  const int x0   = blockIdx.x * TILE;
  const int tid  = threadIdx.x;
  const int lane = tid & 31;
  const int wave = __builtin_amdgcn_readfirstlane(tid >> 5);  // scalar wave id (8 waves)
  const int half = (lane >> 4) & 1;
  const int l15  = lane & 15;

  // A matrix: 16 ref pixels (M on lanes) x 32 channels (K on elements).
  // k order used here (same as B): lane L -> M=L%16; elem e -> K = (L>=16?16:0) + e
  const __bf16* refp =
      featbf + ((size_t)((b * VV + 0) * HH + y) * WW + (x0 + l15)) * CC;
  v16bf Am;
  {
    v8bf lo = *(const v8bf*)(refp + half * 16);       // channels half*16 .. +7
    v8bf hi = *(const v8bf*)(refp + half * 16 + 8);   // channels half*16+8 .. +15
#pragma unroll
    for (int e = 0; e < 8; ++e) { Am[e] = lo[e]; Am[8 + e] = hi[e]; }
  }

  float acc[3] = {0.f, 0.f, 0.f};       // per-thread (d,p) accumulators across views
  const int pMine = tid & 15;           // same ref pixel for all 3 items of this thread
  const int xMine = x0 + pMine;

  for (int v = 1; v < VV; ++v) {
    const float* Rp = rt + ((b * NSRC) + (v - 1)) * 12;
    const float R0 = Rp[0], R1 = Rp[1], R2 = Rp[2];
    const float R3 = Rp[3], R4 = Rp[4], R5 = Rp[5];
    const float R6 = Rp[6], R7 = Rp[7], R8 = Rp[8];
    const float T0 = Rp[9], T1 = Rp[10], T2 = Rp[11];

    __syncthreads();                    // protect corr reads of previous view

    // ---- exact bbox: px,py Möbius => extremes at x-corners; scan all 48 depths
    if (tid < 96) {
      int di = tid % DD;
      int xc = (tid < DD) ? x0 : (x0 + TILE - 1);
      float d  = dv[b * DD + di];
      float gx = (float)xc, gy = (float)y;
      float rx = R0 * gx + R1 * gy + R2;
      float ry = R3 * gx + R4 * gy + R5;
      float rz = R6 * gx + R7 * gy + R8;
      float inv = __builtin_amdgcn_rcpf(rz * d + T2);
      sx[tid] = (rx * d + T0) * inv;
      sy[tid] = (ry * d + T1) * inv;
    }
    __syncthreads();
    if (tid < 32) {                     // wave32 shuffle-tree min/max reduction
      float mnx = fminf(sx[tid], fminf(sx[tid + 32], sx[tid + 64]));
      float mxx = fmaxf(sx[tid], fmaxf(sx[tid + 32], sx[tid + 64]));
      float mny = fminf(sy[tid], fminf(sy[tid + 32], sy[tid + 64]));
      float mxy = fmaxf(sy[tid], fmaxf(sy[tid + 32], sy[tid + 64]));
#pragma unroll
      for (int off = 16; off >= 1; off >>= 1) {
        mnx = fminf(mnx, __shfl_xor(mnx, off));
        mxx = fmaxf(mxx, __shfl_xor(mxx, off));
        mny = fminf(mny, __shfl_xor(mny, off));
        mxy = fmaxf(mxy, __shfl_xor(mxy, off));
      }
      if (tid == 0) {
        int qx0 = imax(0, imin((int)floorf(mnx) - 1, WW - 1));
        int qx1 = imax(0, imin((int)floorf(mxx) + 2, WW - 1));
        int qy0 = imax(0, imin((int)floorf(mny) - 1, HH - 1));
        int qy1 = imax(0, imin((int)floorf(mxy) + 2, HH - 1));
        int qw = imax(1, imin(qx1 - qx0 + 1, QW_MAX));
        int qh = imax(1, imin(qy1 - qy0 + 1, QH_MAX));
        int nq = qw * qh;
        sbox[0] = qx0; sbox[1] = qy0; sbox[2] = qw; sbox[3] = qh;
        sbox[4] = nq;  sbox[5] = (nq + 15) >> 4;
      }
    }
    __syncthreads();
    // Force box parameters into SGPRs: loop control goes scalar, EXEC stays all-1s.
    const int qx0    = __builtin_amdgcn_readfirstlane(sbox[0]);
    const int qy0    = __builtin_amdgcn_readfirstlane(sbox[1]);
    const int qw     = __builtin_amdgcn_readfirstlane(sbox[2]);
    const int qh     = __builtin_amdgcn_readfirstlane(sbox[3]);
    const int nq     = __builtin_amdgcn_readfirstlane(sbox[4]);
    const int nchunk = __builtin_amdgcn_readfirstlane(sbox[5]);
    const float rcpqw = __builtin_amdgcn_rcpf((float)qw);

    // ---- corr[nq x 16] = (refTile(16x32) @ srcRegion(32 x nq))^T via WMMA bf16.
    //      Software-pipelined 2x: next chunk's loads issue before the current WMMA.
    const __bf16* srcp = featbf + (size_t)((b * VV + v) * HH) * WW * CC;
    // B layout: 32 channels (K) x 16 q-cols (N on lanes); lanes 0-15 K=0..15, 16-31 K=16..31
    auto baddr = [&](int jj) -> const __bf16* {
      int q = imin(jj * 16 + l15, nq - 1);
      int qy = (int)(((float)q + 0.5f) * rcpqw);   // exact floor(q/qw): margin >> rcp error
      int qx = q - qy * qw;
      return srcp + ((size_t)(qy0 + qy) * WW + (qx0 + qx)) * CC + half * 16;
    };
    auto do_wmma = [&](int jj, v8bf lo, v8bf hi) {
      v16bf Bm;
#pragma unroll
      for (int e = 0; e < 8; ++e) { Bm[e] = lo[e]; Bm[8 + e] = hi[e]; }
      v8f Cm = {0.f, 0.f, 0.f, 0.f, 0.f, 0.f, 0.f, 0.f};
      Cm = __builtin_amdgcn_wmma_f32_16x16x32_bf16(
          false, Am, false, Bm, (short)0, Cm, false, false);
      // C/D layout: VGPR r <-> ref pixel M = r + half*8; column q = jj*16 + lane%16.
      // Transposed corr: lane's 8 results are contiguous -> two ds_store_b128.
      int qcol = jj * 16 + l15;
      v4f lo4 = {Cm[0], Cm[1], Cm[2], Cm[3]};
      v4f hi4 = {Cm[4], Cm[5], Cm[6], Cm[7]};
      *(v4f*)(&corr[qcol][half * 8])     = lo4;
      *(v4f*)(&corr[qcol][half * 8 + 4]) = hi4;
    };

    int j = wave;
    if (j < nchunk) {
      const __bf16* p = baddr(j);
      v8bf c0 = *(const v8bf*)p, c1 = *(const v8bf*)(p + 8);
      for (;;) {
        int j1 = j + 8;
        if (j1 >= nchunk) { do_wmma(j, c0, c1); break; }
        const __bf16* p1 = baddr(j1);
        __builtin_prefetch(p1 + 128 * CC, 0, 3);
        v8bf d0 = *(const v8bf*)p1, d1 = *(const v8bf*)(p1 + 8);
        do_wmma(j, c0, c1);
        int j2 = j1 + 8;
        if (j2 >= nchunk) { do_wmma(j1, d0, d1); break; }
        const __bf16* p2 = baddr(j2);
        __builtin_prefetch(p2 + 128 * CC, 0, 3);
        c0 = *(const v8bf*)p2; c1 = *(const v8bf*)(p2 + 8);
        do_wmma(j1, d0, d1);
        j = j2;
      }
    }
    __syncthreads();

    // ---- blend: out[d,p] = sum_tap w_tap * corr[tap][p]; 768 items / 256 threads
    const float wview = vw[((size_t)(b * NSRC + (v - 1)) * HH + y) * WW + xMine];
    const float gx = (float)xMine, gy = (float)y;
    const float rx = R0 * gx + R1 * gy + R2;
    const float ry = R3 * gx + R4 * gy + R5;
    const float rz = R6 * gx + R7 * gy + R8;
#pragma unroll
    for (int it = 0; it < 3; ++it) {
      int idx = tid + it * 256;
      int d = idx >> 4;
      float depth = dv[b * DD + d];
      float inv = __builtin_amdgcn_rcpf(rz * depth + T2);
      float px = (rx * depth + T0) * inv;
      float py = (ry * depth + T1) * inv;
      float fx = floorf(px), fy = floorf(py);
      float wx = px - fx, wy = py - fy;
      float s = 0.f;
#pragma unroll
      for (int cny = 0; cny < 2; ++cny) {
#pragma unroll
        for (int cnx = 0; cnx < 2; ++cnx) {
          float xi = fx + (float)cnx, yi = fy + (float)cny;
          float wgt = (cnx ? wx : 1.f - wx) * (cny ? wy : 1.f - wy);
          bool valid = (xi >= 0.f) && (xi <= (float)(WW - 1)) &&
                       (yi >= 0.f) && (yi <= (float)(HH - 1));
          int xc = imin(imax((int)xi, 0), WW - 1);
          int yc = imin(imax((int)yi, 0), HH - 1);
          int lqx = imin(imax(xc - qx0, 0), qw - 1);
          int lqy = imin(imax(yc - qy0, 0), qh - 1);
          s += corr[lqy * qw + lqx][pMine] * (valid ? wgt : 0.f);
        }
      }
      acc[it] += s * (1.0f / (float)CC) * wview;
    }
  } // view loop

  // ---- final: divide by (1e-5 + sum of view weights), write (B,1,D,H,W)
  float wsum = 1e-5f;
  for (int v = 0; v < NSRC; ++v)
    wsum += vw[((size_t)(b * NSRC + v) * HH + y) * WW + xMine];
  const float rw = __builtin_amdgcn_rcpf(wsum);
#pragma unroll
  for (int it = 0; it < 3; ++it) {
    int idx = tid + it * 256;
    int d = idx >> 4;
    out[((size_t)(b * DD + d) * HH + y) * WW + xMine] = acc[it] * rw;
  }
}

extern "C" void kernel_launch(void* const* d_in, const int* in_sizes, int n_in,
                              void* d_out, int out_size, void* d_ws, size_t ws_size,
                              hipStream_t stream) {
  const float* feat = (const float*)d_in[0];   // features (B,V,C,H,W)
  const float* pm   = (const float*)d_in[1];   // proj_matrices (B,V,2,4,4)
  const float* dv   = (const float*)d_in[2];   // depth_values (B,D)
  const float* vw   = (const float*)d_in[3];   // view_weights (B,V-1,H,W)
  float*  ws     = (float*)d_ws;               // [0,384B): per-(b,v) rot/trans
  __bf16* featbf = (__bf16*)((char*)d_ws + WS_BF16_OFFSET);  // 12.5MB bf16 copy
  float* outF = (float*)d_out;

  mvs_setup_kernel<<<1, 1, 0, stream>>>(pm, ws);

  const int npix = BB * VV * HW;               // 204800
  mvs_convert_kernel<<<(npix + 255) / 256, 256, 0, stream>>>(feat, featbf);

  dim3 grid(WW / TILE, HH, BB);                // 10 x 128 x 2 = 2560 blocks
  mvs_main_kernel<<<grid, 256, 0, stream>>>(featbf, dv, vw, ws, outF);
}
